// CausalSelfAttention_20315195310122
// MI455X (gfx1250) — matrix operand order repaired and checked by reading, not compiled, not run on hardware
//
#include <hip/hip_runtime.h>

// ---------------------------------------------------------------------------
// CausalSelfAttention for MI455X (gfx1250): bf16 WMMA, flash-style softmax,
// fp32 accumulation. All LDS tiles are fed by the Tensor Data Mover with
// double buffering (TDM overlapped with WMMA compute).
// ---------------------------------------------------------------------------

typedef __attribute__((ext_vector_type(16))) __bf16 bf16x16;
typedef __attribute__((ext_vector_type(8)))  __bf16 bf16x8;
typedef __attribute__((ext_vector_type(8)))  float  floatx8;
typedef __attribute__((ext_vector_type(4)))  unsigned int uint32x4;
typedef __attribute__((ext_vector_type(8)))  int   int32x8;
typedef __attribute__((ext_vector_type(4)))  int   int32x4;

#define BB 4
#define TT 2048
#define CC 1024
#define HH 16
#define HD 64
#define BT (BB * TT)   // 8192 tokens

union FragU { bf16x16 v; bf16x8 h[2]; };

__device__ __forceinline__ floatx8 zero8() {
  floatx8 v;
#pragma unroll
  for (int i = 0; i < 8; i++) v[i] = 0.0f;
  return v;
}

// A fragment (16x32 bf16, row-major source, leading dim `ld` elements).
// Lanes 0-15: row M=lane, K = 0..7,16..23 ; lanes 16-31: same rows, K +8.
__device__ __forceinline__ bf16x16 load_a_frag(const __bf16* src, int ld) {
  int lane = threadIdx.x & 31;
  int row  = lane & 15;
  int kb   = (lane >> 4) << 3;            // 0 or 8
  const __bf16* p = src + row * ld + kb;
  FragU f;
  f.h[0] = *(const bf16x8*)(p);           // K = kb .. kb+7
  f.h[1] = *(const bf16x8*)(p + 16);      // K = kb+16 .. kb+23
  return f.v;
}

// B fragment (32x16 bf16). Source stored K-contiguous per column:
// column n begins at colbase + n*ldn. Lanes 0-15: K=0..7,16..23; 16-31: K +8.
__device__ __forceinline__ bf16x16 load_b_frag(const __bf16* colbase, int ldn) {
  int lane = threadIdx.x & 31;
  int n  = lane & 15;
  int kb = (lane >> 4) << 3;              // 0 or 8
  const __bf16* p = colbase + n * ldn + kb;
  FragU f;
  f.h[0] = *(const bf16x8*)(p);
  f.h[1] = *(const bf16x8*)(p + 16);
  return f.v;
}

__device__ __forceinline__ floatx8 wmma_bf16(bf16x16 a, bf16x16 b, floatx8 c) {
  return __builtin_amdgcn_wmma_f32_16x16x32_bf16(false, a, false, b,
                                                 (short)0, c, false, false);
}

// ---------------------------------------------------------------------------
// Tensor Data Mover: 2-D bf16 tile (tile_d1 rows of tile_d0 elements) from
// global (row stride `stride0` elements) into LDS at byte offset lds_off.
// D# layout per CDNA5 ISA; extra groups zero.
// 6-arg builtin: (uint32x4 g0, int32x8 g1, int32x4, int32x4, int32x8, cpol)
// ---------------------------------------------------------------------------
__device__ __forceinline__ void tdm_load_2d_bf16(unsigned lds_off, const void* gptr,
                                                 unsigned tile_d0, unsigned tile_d1,
                                                 unsigned tensor_d0, unsigned tensor_d1,
                                                 unsigned stride0) {
  unsigned long long ga = (unsigned long long)(uintptr_t)gptr;
  uint32x4 g0;
  g0[0] = 1u;                                              // count=1, user D#
  g0[1] = lds_off;                                         // lds_addr (bytes)
  g0[2] = (unsigned)(ga & 0xFFFFFFFFull);                  // global_addr[31:0]
  g0[3] = (unsigned)((ga >> 32) & 0x01FFFFFFull) | (2u << 30); // addr[56:32] | type=2
  int32x8 g1;
  g1[0] = (int)(1u << 16);                                 // data_size=1 (2 bytes)
  g1[1] = (int)((tensor_d0 & 0xFFFFu) << 16);              // tensor_dim0 lo16
  g1[2] = (int)(((tensor_d0 >> 16) & 0xFFFFu) |
                ((tensor_d1 & 0xFFFFu) << 16));            // dim0 hi | dim1 lo
  g1[3] = (int)(((tensor_d1 >> 16) & 0xFFFFu) |
                ((tile_d0 & 0xFFFFu) << 16));              // dim1 hi | tile_dim0
  g1[4] = (int)(tile_d1 & 0xFFFFu);                        // tile_dim1, tile_dim2=0
  g1[5] = (int)stride0;                                    // tensor_dim0_stride lo32
  g1[6] = 0;
  g1[7] = 0;
  int32x4 z4;
  z4[0] = 0; z4[1] = 0; z4[2] = 0; z4[3] = 0;
  int32x8 z8;
#pragma unroll
  for (int i = 0; i < 8; i++) z8[i] = 0;
  __builtin_amdgcn_tensor_load_to_lds(g0, g1, z4, z4, z8, 0);
}

// ---------------------------------------------------------------------------
// fp32 -> bf16 conversion (plain, and transposing for weights: out = W^T)
// ---------------------------------------------------------------------------
__global__ void cvt_f32_bf16(const float* __restrict__ in,
                             __bf16* __restrict__ out, int n) {
  int i = blockIdx.x * blockDim.x + threadIdx.x;
  int stride = gridDim.x * blockDim.x;
  for (; i < n; i += stride) out[i] = (__bf16)in[i];
}

__global__ void cvt_t_f32_bf16(const float* __restrict__ in,   // [1024][1024]
                               __bf16* __restrict__ out) {     // out[n][k] = in[k][n]
  int idx = blockIdx.x * blockDim.x + threadIdx.x;   // 0 .. 1024*1024-1
  int k = idx >> 10;
  int n = idx & 1023;
  out[(size_t)n * CC + k] = (__bf16)in[(size_t)k * CC + n];   // coalesced reads
}

// ---------------------------------------------------------------------------
// 128x128x(K=1024) bf16 GEMM, BK=32, 256 threads = 8 waves (4x2 wave grid),
// each wave computes a 32x64 tile = 2x4 WMMA accumulators.
// Both tiles arrive via TDM (A row-major, Wt = W^T row-major) with double
// buffering: next tile's TDM overlaps current tile's WMMAs.
//   mode 0: QKV epilogue -> bf16 head layout [B,H,T,HD]; optional fp32 present
//   mode 2: V epilogue   -> bf16 transposed head layout [B,H,HD,T] + present
//   mode 1: plain fp32 [M,1024] output (final projection)
// ---------------------------------------------------------------------------
__global__ __launch_bounds__(256)
void gemm_bf16_128(const __bf16* __restrict__ A,    // [8192][1024] row-major
                   const __bf16* __restrict__ Wt,   // [1024][1024] = W^T row-major
                   const float*  __restrict__ bias, // [1024]
                   int mode,
                   __bf16* __restrict__ out_bf16,   // mode 0/2
                   float*  __restrict__ out_f32)    // mode 0/2: present; mode 1: y
{
  __shared__ __bf16 ldsA[2][128 * 32];   // [row][k]
  __shared__ __bf16 ldsB[2][128 * 32];   // [n][k]  (K-contiguous per column)

  const int tid  = threadIdx.x;
  const int wave = tid >> 5;
  const int lane = tid & 31;
  const int wr   = wave & 3;          // wave row  0..3  (32 rows each)
  const int wc   = wave >> 2;         // wave col  0..1  (64 cols each)
  const int m0   = blockIdx.y * 128;
  const int n0   = blockIdx.x * 128;

  const unsigned offA[2] = { (unsigned)(uintptr_t)&ldsA[0][0],
                             (unsigned)(uintptr_t)&ldsA[1][0] };
  const unsigned offB[2] = { (unsigned)(uintptr_t)&ldsB[0][0],
                             (unsigned)(uintptr_t)&ldsB[1][0] };
  const __bf16* Abase = A  + (size_t)m0 * 1024;
  const __bf16* Bbase = Wt + (size_t)n0 * 1024;

  floatx8 acc[2][4];
#pragma unroll
  for (int im = 0; im < 2; im++)
#pragma unroll
    for (int in = 0; in < 4; in++) acc[im][in] = zero8();

  // Prologue: fetch first tiles.
  if (wave == 0) {
    tdm_load_2d_bf16(offA[0], Abase, 32, 128, 1024, 8192, 1024);
    tdm_load_2d_bf16(offB[0], Bbase, 32, 128, 1024, 1024, 1024);
    __builtin_amdgcn_s_wait_tensorcnt(0);
  }
  __syncthreads();

  for (int it = 0; it < 32; it++) {
    const int cur = it & 1;
    const int nxt = cur ^ 1;
    // Kick off next tiles; they land in the other buffer while we compute.
    if (wave == 0 && it + 1 < 32) {
      tdm_load_2d_bf16(offA[nxt], Abase + (it + 1) * 32, 32, 128, 1024, 8192, 1024);
      tdm_load_2d_bf16(offB[nxt], Bbase + (it + 1) * 32, 32, 128, 1024, 1024, 1024);
    }

    bf16x16 af[2], bfr[4];
#pragma unroll
    for (int im = 0; im < 2; im++)
      af[im] = load_a_frag(&ldsA[cur][(wr * 32 + im * 16) * 32], 32);
#pragma unroll
    for (int in = 0; in < 4; in++)
      bfr[in] = load_b_frag(&ldsB[cur][(wc * 64 + in * 16) * 32], 32);
#pragma unroll
    for (int im = 0; im < 2; im++)
#pragma unroll
      for (int in = 0; in < 4; in++)
        acc[im][in] = wmma_bf16(af[im], bfr[in], acc[im][in]);

    if (wave == 0) __builtin_amdgcn_s_wait_tensorcnt(0);
    __syncthreads();   // next buffer ready; everyone done reading cur
  }

  // ---- epilogue ----
  const int rowHi = (lane >> 4) << 3;   // 0 or 8
  const int colN  = lane & 15;
#pragma unroll
  for (int im = 0; im < 2; im++) {
#pragma unroll
    for (int in = 0; in < 4; in++) {
      int gcol = n0 + wc * 64 + in * 16 + colN;
      float bval = bias[gcol];
#pragma unroll
      for (int r = 0; r < 8; r++) {
        int grow  = m0 + wr * 32 + im * 16 + rowHi + r;   // token index
        float val = acc[im][in][r] + bval;
        if (mode == 1) {
          out_f32[(size_t)grow * CC + gcol] = val;
        } else {
          int b = grow >> 11;          // / 2048
          int t = grow & 2047;
          int h = gcol >> 6;
          int d = gcol & 63;
          size_t hidx = (((size_t)b * HH + h) * TT + t) * HD + d;
          if (out_f32) out_f32[hidx] = val;            // present (K or V), fp32
          if (mode == 0)
            out_bf16[hidx] = (__bf16)val;              // [B,H,T,HD]
          else
            out_bf16[(((size_t)b * HH + h) * HD + d) * TT + t] = (__bf16)val; // [B,H,HD,T]
        }
      }
    }
  }
}

// ---------------------------------------------------------------------------
// Flash attention: one block = 64 query rows of one (b,h); 4 waves x 16 rows.
// K tile ([key][d], contiguous) and Vt tile ([d][key], stride T) both arrive
// via TDM, double buffered across key tiles.
// ---------------------------------------------------------------------------
__global__ __launch_bounds__(128)
void attn_fwd(const __bf16* __restrict__ qh, const __bf16* __restrict__ kh,
              const __bf16* __restrict__ vt, __bf16* __restrict__ ob)
{
  __shared__ __bf16 ldsK[2][64 * 64];    // [key][d]  row-major
  __shared__ __bf16 ldsVt[2][64 * 64];   // [d][key]
  __shared__ __bf16 ldsP[4][16 * 64];    // per-wave P staging

  const int bh    = blockIdx.x;          // 0..63
  const int b     = bh >> 4;
  const int h     = bh & 15;
  const int i0    = blockIdx.y * 64;     // query tile start
  const int wave  = threadIdx.x >> 5;
  const int lane  = threadIdx.x & 31;
  const int rowHi = (lane >> 4) << 3;    // 0 or 8
  const int colN  = lane & 15;

  const unsigned offK[2]  = { (unsigned)(uintptr_t)&ldsK[0][0],
                              (unsigned)(uintptr_t)&ldsK[1][0] };
  const unsigned offVt[2] = { (unsigned)(uintptr_t)&ldsVt[0][0],
                              (unsigned)(uintptr_t)&ldsVt[1][0] };
  const __bf16* Kbase  = kh + (size_t)bh * TT * HD;   // rows = keys
  const __bf16* VtBase = vt + (size_t)bh * HD * TT;   // rows = d, stride T

  // Q fragments for this wave's 16 rows (HD=64 -> two K=32 chunks), kept live.
  const __bf16* qbase = qh + ((size_t)bh * TT + i0 + wave * 16) * HD;
  bf16x16 qa0 = load_a_frag(qbase, HD);
  bf16x16 qa1 = load_a_frag(qbase + 32, HD);

  floatx8 acc[4];
#pragma unroll
  for (int t = 0; t < 4; t++) acc[t] = zero8();
  float m_s[8], l_s[8];
#pragma unroll
  for (int r = 0; r < 8; r++) { m_s[r] = -__builtin_inff(); l_s[r] = 0.0f; }

  const int ntiles = (i0 >> 6) + 1;      // causal: key tiles 0 .. i0/64

  if (wave == 0) {
    tdm_load_2d_bf16(offK[0],  Kbase,  4096, 1, 4096, 1, 4096);      // 8KB linear
    tdm_load_2d_bf16(offVt[0], VtBase, 64, 64, 2048, 64, 2048);      // [d][j] tile
    __builtin_amdgcn_s_wait_tensorcnt(0);
  }
  __syncthreads();

  for (int jt = 0; jt < ntiles; jt++) {
    const int j0  = jt * 64;
    const int cur = jt & 1;
    const int nxt = cur ^ 1;
    if (wave == 0 && jt + 1 < ntiles) {
      tdm_load_2d_bf16(offK[nxt],  Kbase + (size_t)(j0 + 64) * HD, 4096, 1, 4096, 1, 4096);
      tdm_load_2d_bf16(offVt[nxt], VtBase + (j0 + 64), 64, 64, 2048, 64, 2048);
    }

    // S = Q K^T : 4 n-tiles of 16 keys, 2 WMMAs (d=0..31, 32..63) each.
    floatx8 s[4];
#pragma unroll
    for (int nt = 0; nt < 4; nt++) {
      bf16x16 b0 = load_b_frag(&ldsK[cur][(nt * 16) * 64 + 0],  64);
      bf16x16 b1 = load_b_frag(&ldsK[cur][(nt * 16) * 64 + 32], 64);
      floatx8 c = zero8();
      c = wmma_bf16(qa0, b0, c);
      c = wmma_bf16(qa1, b1, c);
      s[nt] = c;
    }

    // Scale, causal mask, online softmax update (per-row over 16-lane halves).
#pragma unroll
    for (int r = 0; r < 8; r++) {
      int i = i0 + wave * 16 + rowHi + r;
      float rm = -__builtin_inff();
#pragma unroll
      for (int nt = 0; nt < 4; nt++) {
        int j = j0 + nt * 16 + colN;
        float v = s[nt][r] * 0.125f;              // 1/sqrt(64)
        if (j > i) v = -__builtin_inff();
        s[nt][r] = v;
        rm = fmaxf(rm, v);
      }
      rm = fmaxf(rm, __shfl_xor(rm, 1));
      rm = fmaxf(rm, __shfl_xor(rm, 2));
      rm = fmaxf(rm, __shfl_xor(rm, 4));
      rm = fmaxf(rm, __shfl_xor(rm, 8));
      float mn    = fmaxf(m_s[r], rm);
      float alpha = __expf(m_s[r] - mn);          // 0 on first tile (exp(-inf))
      float rsum  = 0.0f;
#pragma unroll
      for (int nt = 0; nt < 4; nt++) {
        float p = __expf(s[nt][r] - mn);
        s[nt][r] = p;
        rsum += p;
      }
      rsum += __shfl_xor(rsum, 1);
      rsum += __shfl_xor(rsum, 2);
      rsum += __shfl_xor(rsum, 4);
      rsum += __shfl_xor(rsum, 8);
      l_s[r] = l_s[r] * alpha + rsum;
      m_s[r] = mn;
#pragma unroll
      for (int nt = 0; nt < 4; nt++) acc[nt][r] = acc[nt][r] * alpha;
    }

    // P: C-layout -> A-fragment layout via per-wave LDS staging (bf16).
#pragma unroll
    for (int nt = 0; nt < 4; nt++)
#pragma unroll
      for (int r = 0; r < 8; r++)
        ldsP[wave][(rowHi + r) * 64 + nt * 16 + colN] = (__bf16)s[nt][r];
    __syncthreads();

    // O += P V : 4 d-tiles x 2 key-chunk WMMAs.
    bf16x16 pa0 = load_a_frag(&ldsP[wave][0],  64);
    bf16x16 pa1 = load_a_frag(&ldsP[wave][32], 64);
#pragma unroll
    for (int nt = 0; nt < 4; nt++) {
      bf16x16 b0 = load_b_frag(&ldsVt[cur][(nt * 16) * 64 + 0],  64);
      bf16x16 b1 = load_b_frag(&ldsVt[cur][(nt * 16) * 64 + 32], 64);
      acc[nt] = wmma_bf16(pa0, b0, acc[nt]);
      acc[nt] = wmma_bf16(pa1, b1, acc[nt]);
    }

    if (wave == 0) __builtin_amdgcn_s_wait_tensorcnt(0);
    __syncthreads();   // next K/Vt buffer ready; everyone done with cur
  }

  // Normalize and write token-layout bf16 for the final projection GEMM.
#pragma unroll
  for (int r = 0; r < 8; r++) {
    float inv = 1.0f / l_s[r];
    int t = i0 + wave * 16 + rowHi + r;
#pragma unroll
    for (int nt = 0; nt < 4; nt++) {
      int c = h * HD + nt * 16 + colN;
      ob[((size_t)b * TT + t) * CC + c] = (__bf16)(acc[nt][r] * inv);
    }
  }
}

// ---------------------------------------------------------------------------
// Host launcher
// ---------------------------------------------------------------------------
extern "C" void kernel_launch(void* const* d_in, const int* in_sizes, int n_in,
                              void* d_out, int out_size, void* d_ws, size_t ws_size,
                              hipStream_t stream) {
  const float* x  = (const float*)d_in[0];
  // d_in[1] = valid_mask: all-True in this problem; causality handled exactly.
  const float* Wq = (const float*)d_in[2];
  const float* bq = (const float*)d_in[3];
  const float* Wk = (const float*)d_in[4];
  const float* bk = (const float*)d_in[5];
  const float* Wv = (const float*)d_in[6];
  const float* bv = (const float*)d_in[7];
  const float* Wp = (const float*)d_in[8];
  const float* bp = (const float*)d_in[9];

  float* y  = (float*)d_out;                       // [B,T,C]
  float* pk = y  + (size_t)BT * CC;                // present[0] = K [B,H,T,HD]
  float* pv = pk + (size_t)BB * HH * TT * HD;      // present[1] = V

  char* w = (char*)d_ws;
  __bf16* xb  = (__bf16*)w; w += (size_t)BT * CC * 2;   // x in bf16
  __bf16* wqt = (__bf16*)w; w += (size_t)CC * CC * 2;   // W^T in bf16
  __bf16* wkt = (__bf16*)w; w += (size_t)CC * CC * 2;
  __bf16* wvt = (__bf16*)w; w += (size_t)CC * CC * 2;
  __bf16* wpt = (__bf16*)w; w += (size_t)CC * CC * 2;
  __bf16* qh  = (__bf16*)w; w += (size_t)BT * CC * 2;   // Q head layout [B,H,T,HD]
  __bf16* kh  = (__bf16*)w; w += (size_t)BT * CC * 2;   // K head layout [B,H,T,HD]
  __bf16* vt  = (__bf16*)w; w += (size_t)BT * CC * 2;   // V transposed  [B,H,HD,T]
  __bf16* ob  = (__bf16*)w; w += (size_t)BT * CC * 2;   // attn out, token layout

  // 1) fp32 -> bf16 conversions (weights transposed once -> TDM-friendly)
  {
    int nx = BT * CC, th = 256;
    cvt_f32_bf16<<<(nx + th * 8 - 1) / (th * 8), th, 0, stream>>>(x, xb, nx);
    int nwblk = (CC * CC) / th;          // 4096 blocks, 1 elem/thread
    cvt_t_f32_bf16<<<nwblk, th, 0, stream>>>(Wq, wqt);
    cvt_t_f32_bf16<<<nwblk, th, 0, stream>>>(Wk, wkt);
    cvt_t_f32_bf16<<<nwblk, th, 0, stream>>>(Wv, wvt);
    cvt_t_f32_bf16<<<nwblk, th, 0, stream>>>(Wp, wpt);
  }

  // 2) QKV projections (WMMA GEMM); K/V also write fp32 `present`.
  dim3 ggrid(CC / 128, BT / 128);   // (8, 64)
  gemm_bf16_128<<<ggrid, 256, 0, stream>>>(xb, wqt, bq, 0, qh, nullptr);
  gemm_bf16_128<<<ggrid, 256, 0, stream>>>(xb, wkt, bk, 0, kh, pk);
  gemm_bf16_128<<<ggrid, 256, 0, stream>>>(xb, wvt, bv, 2, vt, pv);

  // 3) Flash attention
  dim3 agrid(BB * HH, TT / 64);     // (64, 32)
  attn_fwd<<<agrid, 128, 0, stream>>>(qh, kh, vt, ob);

  // 4) Output projection -> fp32 y
  gemm_bf16_128<<<ggrid, 256, 0, stream>>>(ob, wpt, bp, 1, nullptr, y);
}
